// SigmaMoE_57698590655101
// MI455X (gfx1250) — compile-verified, matrix operand order checked by reading, not run on hardware
//
#include <hip/hip_runtime.h>
#include <hip/hip_bf16.h>

#define NTOK 4096   // B*S
#define DIM  512    // D
#define NEXP 64     // E
#define ESZ  128    // expert hidden size
#define TOPK 4
#define TW   128    // threads per expert workgroup (4 waves)

typedef __attribute__((ext_vector_type(16))) __bf16        v16bf;
typedef __attribute__((ext_vector_type(8)))  float         v8f;
typedef __attribute__((ext_vector_type(4)))  unsigned int  v4u;

union Frag { v4u q[2]; unsigned int u[8]; v16bf v; };

// native __bf16 casts -> hardware cvt (RNE); compiler can fuse to packed cvt
__device__ __forceinline__ unsigned short bf16_1(float a) {
    __bf16 h = (__bf16)a;
    union { __bf16 h; unsigned short u; } c; c.h = h;
    return c.u;
}
__device__ __forceinline__ unsigned int pack_bf16(float a, float b) {
    return (unsigned int)bf16_1(a) | ((unsigned int)bf16_1(b) << 16);
}

// ---------------------------------------------------------------- kernel 0
__global__ void zero_counts(int* cnt) {
    if (threadIdx.x < NEXP) cnt[threadIdx.x] = 0;
}

// ---------------------------------------------------------------- kernel 1
// One 64-thread block per token: router scores + sigmoid + top-4 + bucket append.
__global__ __launch_bounds__(64) void router_topk(
    const float* __restrict__ x, const float* __restrict__ sel_w,
    int* __restrict__ cnt, int* __restrict__ bucket_id, float* __restrict__ bucket_w)
{
    const int n = blockIdx.x;
    const int e = threadIdx.x;
    __shared__ float sc[NEXP];

    const float4* xv = (const float4*)(x + (size_t)n * DIM);
    const float4* wv = (const float4*)(sel_w + (size_t)e * DIM);
    float acc = 0.f;
#pragma unroll 4
    for (int i = 0; i < DIM / 4; ++i) {
        float4 a = xv[i], b = wv[i];
        acc += a.x * b.x + a.y * b.y + a.z * b.z + a.w * b.w;
    }
    sc[e] = 1.0f / (1.0f + __expf(-acc));   // sigmoid (monotone => same top-k)
    __syncthreads();

    if (e == 0) {
#pragma unroll
        for (int k = 0; k < TOPK; ++k) {
            float best = -1.f; int bi = 0;
            for (int j = 0; j < NEXP; ++j) {
                float v = sc[j];
                if (v > best) { best = v; bi = j; }
            }
            sc[bi] = -2.f;
            int pos = atomicAdd(&cnt[bi], 1);
            bucket_id[bi * NTOK + pos] = (n << 2) | k;
            bucket_w[bi * NTOK + pos]  = best;
        }
    }
}

// ---------------------------------------------------------------- kernel 2
// Per (expert, 16-token tile): rows = (w * relu(Xg @ keys[e])) @ values[e]
// -> partial[(token*4+slot)*512 + d]   (deterministic, no float atomics).
__global__ __launch_bounds__(TW) void expert_gemm(
    const float* __restrict__ x, const float* __restrict__ keys,
    const float* __restrict__ values, const int* __restrict__ cnt,
    const int* __restrict__ bucket_id, const float* __restrict__ bucket_w,
    float* __restrict__ partial)
{
    const int e  = blockIdx.x;
    const int m0 = blockIdx.y * 16;
    const int ce = cnt[e];
    if (m0 >= ce) return;

    __shared__ __attribute__((aligned(16))) unsigned short A_sh[16 * DIM]; // bf16 X rows (16 KB)
    __shared__ __attribute__((aligned(16))) unsigned short B_sh[ESZ * 32]; // keys chunk [n][k] (8 KB)
    __shared__ __attribute__((aligned(16))) unsigned short H_sh[16 * ESZ]; // hidden bf16 [m][es] (4 KB)
    __shared__ __attribute__((aligned(16))) unsigned short V_sh[DIM * 32]; // values chunk [n][k] (32 KB)
    __shared__ float w_sh[16];
    __shared__ int   pid_sh[16];
    __shared__ int   tok_sh[16];

    const int tid  = threadIdx.x;
    const int wv   = tid >> 5;        // wave 0..3
    const int ln   = tid & 31;
    const int half = ln >> 4;
    const int l15  = ln & 15;

    if (tid < 16) {
        int m = tid;
        bool valid = (m0 + m) < ce;
        int   pid = valid ? bucket_id[e * NTOK + m0 + m] : 0;
        float w   = valid ? bucket_w[e * NTOK + m0 + m] : 0.f;
        pid_sh[m] = valid ? pid : -1;
        tok_sh[m] = pid >> 2;
        w_sh[m]   = w;
    }
    __syncthreads();

    // ---- stage gathered A rows as bf16 (float4 loads, b64 LDS stores) ----
    {
        unsigned int* A32 = (unsigned int*)A_sh;
        for (int i = tid; i < 16 * (DIM / 4); i += TW) {   // 2048 units
            int m  = i >> 7;            // DIM/4 = 128
            int c4 = i & 127;
            float4 r = {0.f, 0.f, 0.f, 0.f};
            if (pid_sh[m] >= 0)
                r = *(const float4*)(x + (size_t)tok_sh[m] * DIM + c4 * 4);
            uint2 pk = make_uint2(pack_bf16(r.x, r.y), pack_bf16(r.z, r.w));
            *(uint2*)&A32[m * (DIM / 2) + c4 * 2] = pk;
        }
    }

    // ---------------- phase A: H = relu(Xg @ keys[e]) * w  (16 x 128) -----
    v8f acc0 = {}, acc1 = {};
    for (int kk = 0; kk < DIM / 32; ++kk) {
        const float* kp = keys + ((size_t)e * DIM + kk * 32) * ESZ;
        if (kk + 1 < DIM / 32) __builtin_prefetch(kp + 32 * ESZ, 0, 1);
        {   // two adjacent K-rows per step -> packed u32 K-pairs in [n][k] LDS
            unsigned int* B32 = (unsigned int*)B_sh;
            for (int i = tid; i < 16 * 32; i += TW) {      // 512 units
                int kp2 = i >> 5;       // K-pair index 0..15
                int nq  = i & 31;       // n-quad 0..31
                float4 r0 = *(const float4*)(kp + (2 * kp2)     * ESZ + nq * 4);
                float4 r1 = *(const float4*)(kp + (2 * kp2 + 1) * ESZ + nq * 4);
                B32[(nq * 4 + 0) * 16 + kp2] = pack_bf16(r0.x, r1.x);
                B32[(nq * 4 + 1) * 16 + kp2] = pack_bf16(r0.y, r1.y);
                B32[(nq * 4 + 2) * 16 + kp2] = pack_bf16(r0.z, r1.z);
                B32[(nq * 4 + 3) * 16 + kp2] = pack_bf16(r0.w, r1.w);
            }
        }
        __syncthreads();

        Frag A, B0, B1;
        const unsigned short* ar  = &A_sh[l15 * DIM + kk * 32 + half * 8];
        const unsigned short* b0r = &B_sh[(wv * 32 + l15) * 32 + half * 8];
        const unsigned short* b1r = &B_sh[(wv * 32 + 16 + l15) * 32 + half * 8];
        A.q[0]  = *(const v4u*)ar;        A.q[1]  = *(const v4u*)(ar + 16);
        B0.q[0] = *(const v4u*)b0r;       B0.q[1] = *(const v4u*)(b0r + 16);
        B1.q[0] = *(const v4u*)b1r;       B1.q[1] = *(const v4u*)(b1r + 16);

        acc0 = __builtin_amdgcn_wmma_f32_16x16x32_bf16(false, A.v, false, B0.v, (short)0, acc0, false, false);
        acc1 = __builtin_amdgcn_wmma_f32_16x16x32_bf16(false, A.v, false, B1.v, (short)0, acc1, false, false);
        __syncthreads();
    }

    // relu * routing weight, stash H as bf16
#pragma unroll
    for (int r = 0; r < 8; ++r) {
        int m = r + 8 * half;
        float h0 = fmaxf(acc0[r], 0.f) * w_sh[m];
        float h1 = fmaxf(acc1[r], 0.f) * w_sh[m];
        H_sh[m * ESZ + wv * 32 + l15]      = bf16_1(h0);
        H_sh[m * ESZ + wv * 32 + 16 + l15] = bf16_1(h1);
    }
    __syncthreads();

    // ---------------- phase B: out = H @ values[e]  (16 x 512) ------------
    v8f acc[8] = {};
    for (int kc = 0; kc < ESZ / 32; ++kc) {
        const float* vp = values + ((size_t)e * ESZ + kc * 32) * DIM;
        if (kc + 1 < ESZ / 32) __builtin_prefetch(vp + 32 * DIM, 0, 1);
        {
            unsigned int* V32 = (unsigned int*)V_sh;
            for (int i = tid; i < 16 * 128; i += TW) {     // 2048 units
                int kp2 = i >> 7;       // K-pair 0..15
                int nq  = i & 127;      // n-quad 0..127
                float4 r0 = *(const float4*)(vp + (2 * kp2)     * DIM + nq * 4);
                float4 r1 = *(const float4*)(vp + (2 * kp2 + 1) * DIM + nq * 4);
                V32[(nq * 4 + 0) * 16 + kp2] = pack_bf16(r0.x, r1.x);
                V32[(nq * 4 + 1) * 16 + kp2] = pack_bf16(r0.y, r1.y);
                V32[(nq * 4 + 2) * 16 + kp2] = pack_bf16(r0.z, r1.z);
                V32[(nq * 4 + 3) * 16 + kp2] = pack_bf16(r0.w, r1.w);
            }
        }
        __syncthreads();

        Frag A2;
        const unsigned short* hr = &H_sh[l15 * ESZ + kc * 32 + half * 8];
        A2.q[0] = *(const v4u*)hr;  A2.q[1] = *(const v4u*)(hr + 16);
#pragma unroll
        for (int t = 0; t < 8; ++t) {
            int nb = wv * 128 + t * 16;
            Frag Bt;
            const unsigned short* vr = &V_sh[(nb + l15) * 32 + half * 8];
            Bt.q[0] = *(const v4u*)vr;  Bt.q[1] = *(const v4u*)(vr + 16);
            acc[t] = __builtin_amdgcn_wmma_f32_16x16x32_bf16(false, A2.v, false, Bt.v, (short)0, acc[t], false, false);
        }
        __syncthreads();
    }

    // ---- epilogue: transpose through LDS (reuse V_sh) -> coalesced f4 stores
    float* F_sh = (float*)V_sh;                 // 16 x 512 f32 = 32 KB (fits V_sh)
#pragma unroll
    for (int t = 0; t < 8; ++t) {
        int nb = wv * 128 + t * 16 + l15;
#pragma unroll
        for (int r = 0; r < 8; ++r) {
            int m = r + 8 * half;
            F_sh[m * DIM + nb] = acc[t][r];
        }
    }
    __syncthreads();
    for (int i = tid; i < 16 * (DIM / 4); i += TW) {
        int m  = i >> 7;
        int c4 = i & 127;
        int pid = pid_sh[m];
        if (pid >= 0) {
            float4 val = *(const float4*)&F_sh[m * DIM + c4 * 4];
            *(float4*)&partial[(size_t)pid * DIM + c4 * 4] = val;
        }
    }
}

// ---------------------------------------------------------------- kernel 3
__global__ __launch_bounds__(256) void reduce_out(
    const float* __restrict__ partial, float* __restrict__ out)
{
    size_t idx = (size_t)blockIdx.x * 256 + threadIdx.x;   // over NTOK*DIM/4
    size_t n = idx / (DIM / 4), d4 = idx % (DIM / 4);
    const float4* p = (const float4*)(partial + n * (TOPK * DIM)) + d4;
    float4 a = p[0], b = p[DIM / 4], c = p[2 * (DIM / 4)], d = p[3 * (DIM / 4)];
    float4 r;
    r.x = (a.x + b.x) + (c.x + d.x);
    r.y = (a.y + b.y) + (c.y + d.y);
    r.z = (a.z + b.z) + (c.z + d.z);
    r.w = (a.w + b.w) + (c.w + d.w);
    *((float4*)out + idx) = r;
}

// ---------------------------------------------------------------- launch
extern "C" void kernel_launch(void* const* d_in, const int* in_sizes, int n_in,
                              void* d_out, int out_size, void* d_ws, size_t ws_size,
                              hipStream_t stream) {
    const float* x      = (const float*)d_in[0];
    const float* esel   = (const float*)d_in[1];
    const float* keys   = (const float*)d_in[2];
    const float* values = (const float*)d_in[3];
    // d_in[4] = k (always 4 for this problem)

    char*  ws        = (char*)d_ws;
    int*   cnt       = (int*)ws;                                           // 256 B
    int*   bucket_id = (int*)(ws + 1024);                                  // 1 MB
    float* bucket_w  = (float*)(ws + 1024 + (size_t)NEXP * NTOK * 4);      // 1 MB
    float* partial   = (float*)(ws + 1024 + (size_t)NEXP * NTOK * 8);      // 32 MB
    float* out       = (float*)d_out;

    zero_counts<<<1, 64, 0, stream>>>(cnt);
    router_topk<<<NTOK, 64, 0, stream>>>(x, esel, cnt, bucket_id, bucket_w);
    dim3 grid(NEXP, NTOK / 16);
    expert_gemm<<<grid, TW, 0, stream>>>(x, keys, values, cnt, bucket_id, bucket_w, partial);
    reduce_out<<<(NTOK * DIM) / 4 / 256, 256, 0, stream>>>(partial, out);
}
